// ColorResBlock_13280038879815
// MI455X (gfx1250) — matrix operand order repaired and checked by reading, not compiled, not run on hardware
//
#include <hip/hip_runtime.h>
#include <hip/hip_bf16.h>

typedef __attribute__((ext_vector_type(16))) _Float16 v16h;
typedef __attribute__((ext_vector_type(8)))  _Float16 v8h;
typedef __attribute__((ext_vector_type(8)))  float    v8f;
typedef __attribute__((ext_vector_type(4)))  float    v4f;
typedef __attribute__((ext_vector_type(2)))  float    v2f;

#define KH      512
#define NC      6
#define ROWLEN  (KH * NC)          // 3072 floats per (b,pos) row
#define MTILE   16                 // rows per workgroup
#define LROW    (KH + 8)           // 520 halves: +16B pad to spread LDS banks
#define LCH     (MTILE * LROW)     // 8320 halves per channel
#define LDS_BYTES (7 * LCH * 2)    // 116480 B (< 320KB WGP LDS)

// --- tiny pre-pass: conv_weight (512x512x2 f32) -> alpha/beta f16 in ws ---
__global__ void cw_prep(const float* __restrict__ w,
                        _Float16* __restrict__ alphaW,
                        _Float16* __restrict__ betaW) {
    int i = blockIdx.x * blockDim.x + threadIdx.x;   // i = o*512 + k
    if (i < KH * KH) {
        float w0 = w[2 * i], w1 = w[2 * i + 1];
        alphaW[i] = (_Float16)(w0 - w1);
        betaW[i]  = (_Float16)w1;
    }
}

// A operand fragment: two adjacent 16B chunks (16 contiguous k, k = 16*hi + s)
__device__ __forceinline__ v16h loadA(const _Float16* p) {
    v8h a = *(const v8h*)p;
    v8h b = *(const v8h*)(p + 8);
    return __builtin_shufflevector(a, b, 0, 1, 2, 3, 4, 5, 6, 7,
                                         8, 9, 10, 11, 12, 13, 14, 15);
}

// --- fused layernorm + color-conv (7x WMMA GEMM) + bias + relu + residual ---
__global__ __launch_bounds__(256, 1)
void color_resblock(const float* __restrict__ x,
                    const _Float16* __restrict__ alphaW,
                    const _Float16* __restrict__ betaW,
                    const float* __restrict__ convB,
                    const float* __restrict__ gammaW,
                    const float* __restrict__ betaN,
                    float* __restrict__ out) {
    extern __shared__ _Float16 lds[];
    const int tid  = threadIdx.x;
    const int wave = tid >> 5;
    const int lane = tid & 31;
    const int row0 = blockIdx.x * MTILE;

    // ---------------- Phase 1: LayerNorm -> f16 pack into LDS ----------------
    // Each wave handles 2 rows; each lane owns k in [lane*16, lane*16+16),
    // i.e. 96 contiguous floats of the row.
    for (int rr = 0; rr < 2; ++rr) {
        const int  rloc = wave * 2 + rr;                  // 0..15 within tile
        const long row  = row0 + rloc;
        const v4f* src  = (const v4f*)(x + row * (long)ROWLEN + lane * 96);
        v4f v[24];
        float s = 0.f, s2 = 0.f;
        #pragma unroll
        for (int i = 0; i < 24; ++i) {
            v[i] = src[i];
            #pragma unroll
            for (int c = 0; c < 4; ++c) { float f = v[i][c]; s += f; s2 += f * f; }
        }
        #pragma unroll
        for (int off = 16; off > 0; off >>= 1) {
            s  += __shfl_xor(s,  off, 32);
            s2 += __shfl_xor(s2, off, 32);
        }
        const float mean = s * (1.f / ROWLEN);
        const float inv  = rsqrtf(s2 * (1.f / ROWLEN) - mean * mean + 1e-5f);

        #pragma unroll
        for (int jp = 0; jp < 8; ++jp) {                  // pairs of k values
            const int   k  = lane * 16 + jp * 2;
            const float g0 = gammaW[k],     b0 = betaN[k];
            const float g1 = gammaW[k + 1], b1 = betaN[k + 1];
            float f[12];
            #pragma unroll
            for (int q = 0; q < 3; ++q) {
                #pragma unroll
                for (int c = 0; c < 4; ++c) { f[q * 4 + c] = v[jp * 3 + q][c]; }
            }
            float sum0 = 0.f, sum1 = 0.f;
            _Float16 h0[7], h1[7];
            #pragma unroll
            for (int d = 0; d < 6; ++d) {
                float y0 = (f[d]     - mean) * inv * g0 + b0;
                float y1 = (f[6 + d] - mean) * inv * g1 + b1;
                sum0 += y0; sum1 += y1;
                h0[d] = (_Float16)y0; h1[d] = (_Float16)y1;
            }
            h0[6] = (_Float16)sum0; h1[6] = (_Float16)sum1;
            #pragma unroll
            for (int d = 0; d < 7; ++d) {                 // 6 colors + sum chan
                union { _Float16 h[2]; unsigned u; } p;
                p.h[0] = h0[d]; p.h[1] = h1[d];
                *(unsigned*)(&lds[d * LCH + rloc * LROW + k]) = p.u;
            }
        }
    }
    __syncthreads();

    // ---------------- Phase 2: WMMA GEMMs + epilogue ----------------
    const int hi = lane >> 4;      // half-wave select (K-chunk parity)
    const int nl = lane & 15;      // N-lane / M-lane within 16
    #pragma unroll 1
    for (int osub = 0; osub < 4; ++osub) {
        const int o = wave * 64 + osub * 16 + nl;          // output channel
        const _Float16* aWp = alphaW + (long)o * KH + hi * 16;
        const _Float16* bWp = betaW  + (long)o * KH + hi * 16;
        // hoisted per-channel LDS bases (channel offset exceeds DS imm field)
        const _Float16* lch[7];
        #pragma unroll
        for (int c = 0; c < 7; ++c) lch[c] = &lds[c * LCH + nl * LROW + hi * 16];

        v8f acc[7] = {};
        // B operands double-buffered (global/L2 latency); A single-buffered.
        v16h Ba = *(const v16h*)aWp;
        v16h Bb = *(const v16h*)bWp;

        #pragma unroll 2
        for (int kt = 0; kt < 16; ++kt) {
            const int k0 = kt * 32;
            // grouped A loads: 14 ds_load_b128 issued as one batch
            v16h A[7];
            #pragma unroll
            for (int c = 0; c < 7; ++c) A[c] = loadA(lch[c] + k0);
            // prefetch next k-step's B while this step's WMMAs run
            v16h Ban, Bbn;
            if (kt < 15) {
                Ban = *(const v16h*)(aWp + k0 + 32);
                Bbn = *(const v16h*)(bWp + k0 + 32);
            }
            #pragma unroll
            for (int c = 0; c < 6; ++c) {
                acc[c] = __builtin_amdgcn_wmma_f32_16x16x32_f16(
                    false, A[c], false, Ba, (short)0, acc[c], false, false);
            }
            acc[6] = __builtin_amdgcn_wmma_f32_16x16x32_f16(
                false, A[6], false, Bb, (short)0, acc[6], false, false);
            Ba = Ban; Bb = Bbn;   // SSA rotation (renamed away under unroll 2)

#if __has_builtin(__builtin_amdgcn_sched_group_barrier)
            // Pin per-k-step pipeline: 14 DS reads -> 4 global reads -> 7 WMMAs.
            // Keeps all 7 A fragments in flight so the wait inserter can use
            // partial s_wait_dscnt and the WMMAs issue back-to-back.
            __builtin_amdgcn_sched_group_barrier(0x100, 14, 0); // DS read
            __builtin_amdgcn_sched_group_barrier(0x020,  4, 0); // VMEM read
            __builtin_amdgcn_sched_group_barrier(0x008,  7, 0); // MFMA/WMMA
#endif
        }

        // Epilogue: out = x + relu(term1 + term2 + bias)
        const float bo = convB[o];
        #pragma unroll
        for (int j = 0; j < 8; ++j) {
            const int  m    = j + hi * 8;                  // C/D layout: M = vgpr + 8*hi
            const long base = (long)(row0 + m) * ROWLEN + o * NC;
            v2f r0 = *(const v2f*)(x + base);
            v2f r1 = *(const v2f*)(x + base + 2);
            v2f r2 = *(const v2f*)(x + base + 4);
            const float t2 = acc[6][j] + bo;               // term2 + bias
            float y;
            y = acc[0][j] + t2; r0[0] += fmaxf(y, 0.f);
            y = acc[1][j] + t2; r0[1] += fmaxf(y, 0.f);
            y = acc[2][j] + t2; r1[0] += fmaxf(y, 0.f);
            y = acc[3][j] + t2; r1[1] += fmaxf(y, 0.f);
            y = acc[4][j] + t2; r2[0] += fmaxf(y, 0.f);
            y = acc[5][j] + t2; r2[1] += fmaxf(y, 0.f);
            *(v2f*)(out + base)     = r0;
            *(v2f*)(out + base + 2) = r1;
            *(v2f*)(out + base + 4) = r2;
        }
    }
}

extern "C" void kernel_launch(void* const* d_in, const int* in_sizes, int n_in,
                              void* d_out, int out_size, void* d_ws, size_t ws_size,
                              hipStream_t stream) {
    const float* x     = (const float*)d_in[0];   // (1024, 24, 3072)
    const float* convW = (const float*)d_in[1];   // (512, 512, 2)
    const float* convB = (const float*)d_in[2];   // (512,)
    const float* gamma = (const float*)d_in[3];   // (512,)
    const float* nbeta = (const float*)d_in[4];   // (512,)
    float* out = (float*)d_out;

    _Float16* alphaW = (_Float16*)d_ws;           // 512*512 f16
    _Float16* betaW  = alphaW + KH * KH;          // 512*512 f16 (ws: 1 MB total)

    const int rows = in_sizes[0] / ROWLEN;        // 24576

    (void)hipFuncSetAttribute((const void*)color_resblock,
                              hipFuncAttributeMaxDynamicSharedMemorySize,
                              LDS_BYTES);

    cw_prep<<<(KH * KH + 255) / 256, 256, 0, stream>>>(convW, alphaW, betaW);
    color_resblock<<<rows / MTILE, 256, LDS_BYTES, stream>>>(
        x, alphaW, betaW, convB, gamma, nbeta, out);
}
